// Policy_90477781058028
// MI455X (gfx1250) — matrix operand order repaired and checked by reading, not compiled, not run on hardware
//
#include <hip/hip_runtime.h>
#include <hip/hip_bf16.h>
#include <stdint.h>

// Problem sizes (match reference)
#define S_   512
#define H_   2048
#define FH_  8192      // 4*H
#define O_   512
#define T_   2048

// Persistent recurrent kernel geometry: must be co-resident for grid sync.
// 132 blocks x 256 threads = 1056 wave32s == exactly one row-tile per wave in
// phase 1 (512 Whh0 + 512 Whh1 + 32 Wout tiles) -> minimal critical path.
// ~25KB LDS / 256 thr per block; MI455X has >=128 WGPs, residency comfortable.
#define NB_REC 132
#define TPB    256
#define WPB    (TPB / 32)
#define NW_REC (NB_REC * WPB)   // 1056

typedef __bf16 bf16_t;
typedef bf16_t bf16x16 __attribute__((ext_vector_type(16)));
typedef float  f32x8   __attribute__((ext_vector_type(8)));

struct Pack32 { uint4 lo, hi; };   // 32 bytes == v16 bf16

__device__ __forceinline__ unsigned short f2bf(float f) {
  union { float f; unsigned u; } v; v.f = f;
  unsigned r = v.u + 0x7fffu + ((v.u >> 16) & 1u);   // round-to-nearest-even
  return (unsigned short)(r >> 16);
}

// A-operand (16-bit 16x32 MxK layout): lane l(0..15)=row M=l, K in {kA..kA+7, kA+16..kA+23},
// kA = hiHalf*8.  Two contiguous 16B loads.
__device__ __forceinline__ bf16x16 ldA(const unsigned short* __restrict__ row, int k0, int hh) {
  Pack32 p;
  p.lo = *(const uint4*)(row + k0 + hh * 8);
  p.hi = *(const uint4*)(row + k0 + hh * 8 + 16);
  return __builtin_bit_cast(bf16x16, p);
}
// B-operand (16-bit 32x16 KxN): same K order as A: K in {kA..kA+7, kA+16..kA+23}, kA = hiHalf*8,
// column = lane&15.  We broadcast h across all N columns -> two 16B loads per lane.
__device__ __forceinline__ bf16x16 ldB(const unsigned short* h, int k0, int hh) {
  Pack32 p;
  p.lo = *(const uint4*)(h + k0 + hh * 8);
  p.hi = *(const uint4*)(h + k0 + hh * 8 + 16);
  return __builtin_bit_cast(bf16x16, p);
}

// y[r0..r0+16) (+= / =) W[r0.., :H_] * h   via WMMA bf16, C columns replicated.
__device__ __forceinline__ void matvec_tile(const unsigned short* __restrict__ W, int r0,
                                            const unsigned short* h,
                                            float* out, const float* addSrc,
                                            bool accumulate, int lane) {
  const int l15 = lane & 15, hh = lane >> 4;
  const unsigned short* row = W + (size_t)(r0 + l15) * H_;
  f32x8 acc = {0.f, 0.f, 0.f, 0.f, 0.f, 0.f, 0.f, 0.f};
#pragma unroll 4
  for (int k0 = 0; k0 < H_; k0 += 32) {
    __builtin_prefetch(row + k0 + 512, 0, 0);          // global_prefetch_b8 (speculative)
    bf16x16 a = ldA(row, k0, hh);
    bf16x16 b = ldB(h, k0, hh);
    acc = __builtin_amdgcn_wmma_f32_16x16x32_bf16(false, a, false, b, (short)0, acc,
                                                  false, false);
  }
  if (l15 == 0) {                      // lanes 0 and 16 hold rows r0..+7 / r0+8..+15
    int base = r0 + hh * 8;
#pragma unroll
    for (int m = 0; m < 8; ++m) {
      float v = acc[m];
      if (addSrc)     v += addSrc[base + m];
      if (accumulate) v += out[base + m];
      out[base + m] = v;
    }
  }
}

__device__ __forceinline__ void grid_barrier(unsigned* cnt, unsigned* gen, unsigned& barNum) {
  unsigned target = ++barNum;
  __syncthreads();
  if (threadIdx.x == 0) {
    __threadfence();                                   // release: gates visible
    unsigned old = atomicAdd(cnt, 1u);
    if (old == NB_REC - 1) {
      atomicExch(cnt, 0u);
      __threadfence();
      atomicAdd(gen, 1u);
    } else {
      volatile unsigned* vg = (volatile unsigned*)gen;
      while (*vg < target) __builtin_amdgcn_s_sleep(2);
    }
  }
  __syncthreads();
  __threadfence();                                     // acquire
}

// ---------------------------------------------------------------- prep ----
__global__ void prep_kernel(const float* __restrict__ x,    const float* __restrict__ wih0,
                            const float* __restrict__ whh0, const float* __restrict__ bih0,
                            const float* __restrict__ bhh0, const float* __restrict__ wih1,
                            const float* __restrict__ whh1, const float* __restrict__ bih1,
                            const float* __restrict__ bhh1, const float* __restrict__ wout,
                            unsigned short* xbf,   unsigned short* wih0bf,
                            unsigned short* whh0bf, unsigned short* wih1bf,
                            unsigned short* whh1bf, unsigned short* woutbf,
                            float* bias0, float* bias1,
                            unsigned* barCnt, unsigned* barGen) {
  size_t id = (size_t)blockIdx.x * blockDim.x + threadIdx.x;
  size_t st = (size_t)gridDim.x * blockDim.x;
  for (size_t i = id; i < (size_t)FH_ * H_; i += st) {
    whh0bf[i] = f2bf(whh0[i]);
    wih1bf[i] = f2bf(wih1[i]);
    whh1bf[i] = f2bf(whh1[i]);
  }
  for (size_t i = id; i < (size_t)FH_ * S_; i += st) wih0bf[i] = f2bf(wih0[i]);
  for (size_t i = id; i < (size_t)O_ * H_;  i += st) woutbf[i] = f2bf(wout[i]);
  for (size_t i = id; i < (size_t)T_ * S_;  i += st) xbf[i]    = f2bf(x[i]);
  for (size_t i = id; i < (size_t)FH_;      i += st) {
    bias0[i] = bih0[i] + bhh0[i];
    bias1[i] = bih1[i] + bhh1[i];
  }
  if (id == 0) { *barCnt = 0u; *barGen = 0u; }         // replay-safe barrier reset
}

// ------------------------------------------- input projection (big GEMM) ----
// pre[t][j] = sum_k x[t,k]*W_ih0[j,k] + bias0[j]   (2048 x 8192 x 512, WMMA bf16)
__global__ void gemm_pre_kernel(const unsigned short* __restrict__ xbf,
                                const unsigned short* __restrict__ wih0bf,
                                const float* __restrict__ bias0,
                                float* __restrict__ pre) {
  const int lane = threadIdx.x & 31, l15 = lane & 15, hh = lane >> 4;
  int gw = blockIdx.x * WPB + (threadIdx.x >> 5);
  int nw = gridDim.x * WPB;
  const int TT = (T_ / 16) * (FH_ / 16);               // 128 * 512 tiles
  for (int tt = gw; tt < TT; tt += nw) {
    int jt = tt & 511, ti = tt >> 9;
    int t0 = ti * 16, j0 = jt * 16;
    const unsigned short* arow = xbf    + (size_t)(t0 + l15) * S_;
    const unsigned short* brow = wih0bf + (size_t)(j0 + l15) * S_;
    f32x8 acc = {0.f, 0.f, 0.f, 0.f, 0.f, 0.f, 0.f, 0.f};
#pragma unroll 4
    for (int k0 = 0; k0 < S_; k0 += 32) {
      Pack32 pa, pb;
      pa.lo = *(const uint4*)(arow + k0 + hh * 8);
      pa.hi = *(const uint4*)(arow + k0 + hh * 8 + 16);
      pb.lo = *(const uint4*)(brow + k0 + hh * 8);
      pb.hi = *(const uint4*)(brow + k0 + hh * 8 + 16);
      bf16x16 a = __builtin_bit_cast(bf16x16, pa);
      bf16x16 b = __builtin_bit_cast(bf16x16, pb);
      acc = __builtin_amdgcn_wmma_f32_16x16x32_bf16(false, a, false, b, (short)0, acc,
                                                    false, false);
    }
    int col = j0 + l15;
    float bb = bias0[col];
#pragma unroll
    for (int m = 0; m < 8; ++m)
      pre[(size_t)(t0 + hh * 8 + m) * FH_ + col] = acc[m] + bb;
  }
}

// ------------------------------------------------ persistent recurrence ----
__global__ void __launch_bounds__(TPB, 1)
lstm_persistent_kernel(const unsigned short* __restrict__ whh0,
                       const unsigned short* __restrict__ wih1,
                       const unsigned short* __restrict__ whh1,
                       const unsigned short* __restrict__ wout,
                       const float* __restrict__ pre,
                       const float* __restrict__ bias1,
                       const float* __restrict__ bout,
                       float* gates0, float* gates1 /*2*FH_*/, float* logits /*2*O_*/,
                       unsigned* barCnt, unsigned* barGen,
                       float* __restrict__ out) {
  __shared__ __align__(32) float          sC0[H_], sC1[H_];
  __shared__ __align__(32) unsigned short sH0[H_], sH1[H_];
  __shared__ float sRed[TPB];
  const int tid  = threadIdx.x;
  const int lane = tid & 31;
  const int gw   = blockIdx.x * WPB + (tid >> 5);
  unsigned barNum = 0;

  for (int j = tid; j < H_; j += TPB) { sC0[j] = 0.f; sC1[j] = 0.f; sH0[j] = 0; sH1[j] = 0; }
  __syncthreads();

  for (int t = 0; t <= T_; ++t) {
    // Phase 1: gates0 = pre[t] + Whh0*h0 ; gates1_partial = Whh1*h1 ; logits(t-1) = Wout*h1
    float* g1 = gates1 + (size_t)(t & 1) * FH_;
    float* lg = logits + (size_t)(t & 1) * O_;
    {
      const int tile = gw;               // NW_REC == 1056: exactly one tile per wave
      if (tile < 512) {
        if (t < T_) matvec_tile(whh0, tile * 16, sH0, gates0,
                                pre + (size_t)t * FH_, false, lane);
      } else if (tile < 1024) {
        if (t < T_) matvec_tile(whh1, (tile - 512) * 16, sH1, g1, nullptr, false, lane);
      } else {
        if (t > 0)  matvec_tile(wout, (tile - 1024) * 16, sH1, lg, nullptr, false, lane);
      }
    }
    grid_barrier(barCnt, barGen, barNum);

    // Block 0: softmax of step t-1 (double-buffered logits -> no extra barrier)
    if (t > 0 && blockIdx.x == 0) {
      float v0 = lg[tid] + bout[tid];
      float v1 = lg[tid + 256] + bout[tid + 256];
      sRed[tid] = fmaxf(v0, v1); __syncthreads();
      for (int s = 128; s > 0; s >>= 1) {
        if (tid < s) sRed[tid] = fmaxf(sRed[tid], sRed[tid + s]);
        __syncthreads();
      }
      float mx = sRed[0]; __syncthreads();
      float e0 = __expf(v0 - mx), e1 = __expf(v1 - mx);
      sRed[tid] = e0 + e1; __syncthreads();
      for (int s = 128; s > 0; s >>= 1) {
        if (tid < s) sRed[tid] += sRed[tid + s];
        __syncthreads();
      }
      float inv = 1.f / sRed[0]; __syncthreads();
      out[(size_t)(t - 1) * O_ + tid]       = e0 * inv;
      out[(size_t)(t - 1) * O_ + tid + 256] = e1 * inv;
    }
    if (t == T_) break;

    // Cell 0, computed redundantly per block into block-local LDS state.
    for (int j = tid; j < H_; j += TPB) {
      float gi = gates0[j], gf = gates0[H_ + j], gg = gates0[2 * H_ + j], go = gates0[3 * H_ + j];
      float i = 1.f / (1.f + __expf(-gi));
      float f = 1.f / (1.f + __expf(-gf));
      float g = tanhf(gg);
      float o = 1.f / (1.f + __expf(-go));
      float c = f * sC0[j] + i * g;
      sC0[j] = c;
      sH0[j] = f2bf(o * tanhf(c));
    }
    __syncthreads();

    // Phase 2: gates1 += Wih1 * h0  (512 tiles; waves 512.. idle)
    if (gw < 512)
      matvec_tile(wih1, gw * 16, sH0, g1, nullptr, true, lane);
    grid_barrier(barCnt, barGen, barNum);

    // Cell 1, redundant per block (bias1 applied here).
    for (int j = tid; j < H_; j += TPB) {
      float gi = g1[j]          + bias1[j];
      float gf = g1[H_ + j]     + bias1[H_ + j];
      float gg = g1[2 * H_ + j] + bias1[2 * H_ + j];
      float go = g1[3 * H_ + j] + bias1[3 * H_ + j];
      float i = 1.f / (1.f + __expf(-gi));
      float f = 1.f / (1.f + __expf(-gf));
      float g = tanhf(gg);
      float o = 1.f / (1.f + __expf(-go));
      float c = f * sC1[j] + i * g;
      sC1[j] = c;
      sH1[j] = f2bf(o * tanhf(c));
    }
    __syncthreads();
  }
}

// ---------------------------------------------------------------- launch ----
extern "C" void kernel_launch(void* const* d_in, const int* in_sizes, int n_in,
                              void* d_out, int out_size, void* d_ws, size_t ws_size,
                              hipStream_t stream) {
  (void)in_sizes; (void)n_in; (void)out_size; (void)ws_size;
  const float* x    = (const float*)d_in[0];
  const float* wih0 = (const float*)d_in[1];
  const float* whh0 = (const float*)d_in[2];
  const float* bih0 = (const float*)d_in[3];
  const float* bhh0 = (const float*)d_in[4];
  const float* wih1 = (const float*)d_in[5];
  const float* whh1 = (const float*)d_in[6];
  const float* bih1 = (const float*)d_in[7];
  const float* bhh1 = (const float*)d_in[8];
  const float* wout = (const float*)d_in[9];
  const float* bout = (const float*)d_in[10];
  float* out = (float*)d_out;

  char* ws = (char*)d_ws;
  size_t off = 0;
  auto take = [&](size_t bytes) { size_t r = off; off += (bytes + 255) & ~(size_t)255; return r; };
  unsigned short* whh0bf = (unsigned short*)(ws + take((size_t)FH_ * H_ * 2));
  unsigned short* wih1bf = (unsigned short*)(ws + take((size_t)FH_ * H_ * 2));
  unsigned short* whh1bf = (unsigned short*)(ws + take((size_t)FH_ * H_ * 2));
  unsigned short* woutbf = (unsigned short*)(ws + take((size_t)O_ * H_ * 2));
  unsigned short* wih0bf = (unsigned short*)(ws + take((size_t)FH_ * S_ * 2));
  unsigned short* xbf    = (unsigned short*)(ws + take((size_t)T_ * S_ * 2));
  float* bias0  = (float*)(ws + take((size_t)FH_ * 4));
  float* bias1  = (float*)(ws + take((size_t)FH_ * 4));
  float* pre    = (float*)(ws + take((size_t)T_ * FH_ * 4));
  float* gates0 = (float*)(ws + take((size_t)FH_ * 4));
  float* gates1 = (float*)(ws + take((size_t)2 * FH_ * 4));
  float* logits = (float*)(ws + take((size_t)2 * O_ * 4));
  unsigned* barCnt = (unsigned*)(ws + take(256));
  unsigned* barGen = barCnt + 1;

  prep_kernel<<<2048, 256, 0, stream>>>(x, wih0, whh0, bih0, bhh0, wih1, whh1, bih1, bhh1,
                                        wout, xbf, wih0bf, whh0bf, wih1bf, whh1bf, woutbf,
                                        bias0, bias1, barCnt, barGen);
  gemm_pre_kernel<<<1024, 256, 0, stream>>>(xbf, wih0bf, bias0, pre);
  lstm_persistent_kernel<<<NB_REC, TPB, 0, stream>>>(whh0bf, wih1bf, whh1bf, woutbf, pre,
                                                     bias1, bout, gates0, gates1, logits,
                                                     barCnt, barGen, out);
}
